// System2Reasoner_34041910788242
// MI455X (gfx1250) — compile-verified, matrix operand order checked by reading, not run on hardware
//
#include <hip/hip_runtime.h>

#define P_TOTAL 4096
#define M_TOTAL 65536
#define DIM     768
#define KT_N    24          // 768 / 32
#define TP      16          // patches per workgroup (one WMMA M tile)
#define TNF     256         // fast path: nodes per chunk (8 waves x 2 x 16)
#define TN      128         // fallback: nodes per chunk (8 waves x 16)
#define CAND    384         // candidate buffer entries per patch
#define TOPK    50
#define TAU     0.02f
#define NEG_BIG (-3.0e38f)

typedef __attribute__((ext_vector_type(16))) __bf16          v16bf;
typedef __attribute__((ext_vector_type(8)))  float           v8f;
typedef __attribute__((ext_vector_type(16))) unsigned short  u16x16;

__device__ __forceinline__ unsigned short f2bf(float f) {
  unsigned u = __float_as_uint(f);
  u += 0x7FFFu + ((u >> 16) & 1u);          // round-to-nearest-even
  return (unsigned short)(u >> 16);
}
__device__ __forceinline__ float bf2f(unsigned short h) {
  return __uint_as_float(((unsigned)h) << 16);
}

// Wave-wide exact top-50 extraction from cv[0..n) into sv/si (sorted descending).
// Consumed entries are invalidated with NEG_BIG. Must be called wave-uniformly.
__device__ __forceinline__ void extract_top50(float* cv, int* ci,
                                              float* sv, int* si,
                                              int n, int lane) {
  for (int j = 0; j < TOPK; ++j) {
    float best = NEG_BIG; int bpos = 0;
    for (int t = lane; t < n; t += 32) {
      float v = cv[t];
      if (v > best) { best = v; bpos = t; }
    }
    float gmax = best;
    #pragma unroll
    for (int o = 16; o > 0; o >>= 1)
      gmax = fmaxf(gmax, __shfl_xor(gmax, o, 32));
    unsigned long long bal = __ballot(best == gmax);
    int src  = (int)__ffsll(bal) - 1;
    int wpos = __shfl(bpos, src, 32);
    if (lane == 0)  { sv[j] = gmax; si[j] = ci[wpos]; }
    if (lane == src) cv[bpos] = NEG_BIG;
  }
}

// ---------------------------------------------------------------------------
// K0: one-time conversion of memory_nodes f32 -> bf16 hi/lo, stored directly
// in WMMA B-fragment layout:  entry[(nt*24 + kt)*32 + lane] = 16 halves.
// Element (node m, dim k): nt=m>>4, kt=k>>5, kk=k&31:
//   lane = (m&15) + ((kk & 8) ? 16 : 0)
//   v    = ((kk & 16) ? 4 : 0) + ((kk & 7) >> 1),  pos = kk & 1,  e = 2v+pos
// ---------------------------------------------------------------------------
__global__ void __launch_bounds__(256)
s2r_convert_b(const float* __restrict__ B,
              unsigned int* __restrict__ BhiW,    // packed pairs of halves
              unsigned int* __restrict__ BloW) {
  const int nt = blockIdx.x;                 // node tile 0..4095
  const int t  = threadIdx.x;
  const int r  = t >> 4;                     // node row in tile 0..15
  const int c0 = (t & 15) * 2;               // even K within 32-tile
  const int lane = r + ((c0 & 8) ? 16 : 0);
  const int v    = ((c0 & 16) ? 4 : 0) + ((c0 & 7) >> 1);
  const size_t rowbase = (size_t)(nt * 16 + r) * DIM;
  for (int kt = 0; kt < KT_N; ++kt) {
    float2 f = *(const float2*)&B[rowbase + kt * 32 + c0];
    unsigned short h0 = f2bf(f.x), h1 = f2bf(f.y);
    unsigned short l0 = f2bf(f.x - bf2f(h0)), l1 = f2bf(f.y - bf2f(h1));
    size_t entry = ((size_t)nt * KT_N + kt) * 32 + lane;   // 32B fragments
    BhiW[entry * 8 + v] = (unsigned)h0 | ((unsigned)h1 << 16);
    BloW[entry * 8 + v] = (unsigned)l0 | ((unsigned)l1 << 16);
  }
}

// ---------------------------------------------------------------------------
// K1 (fast): GEMM + streaming top-k with preconverted fragment-major B.
// Each wave owns two 16-node subtiles (tiles w and w+8 of a 256-node chunk):
// per K-step 4x ds_load_b128 (A hi/lo) feed 6 WMMAs over two independent
// accumulator chains; all fragment loads use one base pointer per chunk plus
// compile-time immediate offsets (kt*1024, +196608 for the second subtile).
// ---------------------------------------------------------------------------
__global__ void __launch_bounds__(256)
s2r_gemm_topk_fast(const float* __restrict__ A,        // test_patches
                   const char* __restrict__ BhiB,      // fragment-major bytes
                   const char* __restrict__ BloB,
                   float* __restrict__ topval,
                   int*   __restrict__ topidx) {
  __shared__ alignas(32) unsigned short Afrag[2][KT_N][512];   // 48 KB
  __shared__ float thresh[TP];
  __shared__ int   cnt[TP];
  __shared__ float cval[TP][CAND];
  __shared__ int   cidx[TP][CAND];
  __shared__ float selv[TP][TOPK];
  __shared__ int   seli[TP][TOPK];

  const int t    = threadIdx.x;
  const int lane = t & 31;
  const int w    = t >> 5;
  const int p0   = blockIdx.x * TP;

  if (t < TP) { thresh[t] = NEG_BIG; cnt[t] = 0; }

  // A tile -> hi/lo bf16 fragments in LDS
  {
    const int e0    = w * 2;
    const int row   = lane & 15;
    const int klane = (lane & 16) ? 8 : 0;
    for (int kt = 0; kt < KT_N; ++kt) {
      #pragma unroll
      for (int ee = 0; ee < 2; ++ee) {
        int e = e0 + ee;
        int v = e >> 1, pos = e & 1;
        int K = ((v & 4) ? 16 : 0) + klane + ((v & 3) << 1) + pos;
        float f = A[(size_t)(p0 + row) * DIM + kt * 32 + K];
        unsigned short hi = f2bf(f);
        unsigned short lo = f2bf(f - bf2f(hi));
        Afrag[0][kt][lane * 16 + e] = hi;
        Afrag[1][kt][lane * 16 + e] = lo;
      }
    }
  }
  __syncthreads();

  const char* abase = (const char*)&Afrag[0][0][0] + lane * 32;

  for (int n0 = 0; n0 < M_TOTAL; n0 += TNF) {
    const int nt0 = (n0 >> 4) + w;          // this wave's first node tile
    const size_t boff = ((size_t)nt0 * KT_N * 32 + lane) * 32;
    const char* bh0 = BhiB + boff;
    const char* bl0 = BloB + boff;
    v8f acc0 = {}, acc1 = {};

    #pragma unroll 4
    for (int kt = 0; kt < KT_N; ++kt) {
      const int ko = kt * 1024;             // 1 KB per (kt, wave) fragment row
      u16x16 ah16  = *(const u16x16*)(abase + ko);
      u16x16 al16  = *(const u16x16*)(abase + KT_N * 1024 + ko);
      u16x16 bh0v  = *(const u16x16*)(bh0 + ko);
      u16x16 bl0v  = *(const u16x16*)(bl0 + ko);
      u16x16 bh1v  = *(const u16x16*)(bh0 + 8 * KT_N * 1024 + ko);   // tile w+8
      u16x16 bl1v  = *(const u16x16*)(bl0 + 8 * KT_N * 1024 + ko);
      v16bf ah = __builtin_bit_cast(v16bf, ah16);
      v16bf al = __builtin_bit_cast(v16bf, al16);
      v16bf b0h = __builtin_bit_cast(v16bf, bh0v);
      v16bf b0l = __builtin_bit_cast(v16bf, bl0v);
      v16bf b1h = __builtin_bit_cast(v16bf, bh1v);
      v16bf b1l = __builtin_bit_cast(v16bf, bl1v);
      // interleave the two independent accumulation chains
      acc0 = __builtin_amdgcn_wmma_f32_16x16x32_bf16(false, ah, false, b0h, (short)0, acc0, false, false);
      acc1 = __builtin_amdgcn_wmma_f32_16x16x32_bf16(false, ah, false, b1h, (short)0, acc1, false, false);
      acc0 = __builtin_amdgcn_wmma_f32_16x16x32_bf16(false, ah, false, b0l, (short)0, acc0, false, false);
      acc1 = __builtin_amdgcn_wmma_f32_16x16x32_bf16(false, ah, false, b1l, (short)0, acc1, false, false);
      acc0 = __builtin_amdgcn_wmma_f32_16x16x32_bf16(false, al, false, b0h, (short)0, acc0, false, false);
      acc1 = __builtin_amdgcn_wmma_f32_16x16x32_bf16(false, al, false, b1h, (short)0, acc1, false, false);
    }

    const int nodeA = n0 + (w << 4) + (lane & 15);
    const int nodeB = nodeA + 128;          // tile w+8
    const int pbase = (lane & 16) ? 8 : 0;
    float th[8];
    #pragma unroll
    for (int i = 0; i < 8; ++i) th[i] = thresh[pbase + i];
    #pragma unroll
    for (int i = 0; i < 8; ++i) {
      int p = pbase + i;
      float v0 = acc0[i];
      if (v0 > th[i]) {
        int pos = atomicAdd(&cnt[p], 1);
        if (pos < CAND) { cval[p][pos] = v0; cidx[p][pos] = nodeA; }
      }
      float v1 = acc1[i];
      if (v1 > th[i]) {
        int pos = atomicAdd(&cnt[p], 1);
        if (pos < CAND) { cval[p][pos] = v1; cidx[p][pos] = nodeB; }
      }
    }
    __syncthreads();
    #pragma unroll
    for (int o = 0; o < 2; ++o) {
      int p = w + o * 8;
      int c = cnt[p];
      if (c > CAND - TNF) {                 // ensure room for next full chunk
        int n = c < CAND ? c : CAND;
        extract_top50(cval[p], cidx[p], selv[p], seli[p], n, lane);
        for (int j = lane; j < TOPK; j += 32) {
          cval[p][j] = selv[p][j]; cidx[p][j] = seli[p][j];
        }
        if (lane == 0) { cnt[p] = TOPK; thresh[p] = selv[p][TOPK - 1]; }
      }
    }
    __syncthreads();
  }

  #pragma unroll
  for (int o = 0; o < 2; ++o) {
    int p = w + o * 8;
    int c = cnt[p]; int n = c < CAND ? c : CAND;
    extract_top50(cval[p], cidx[p], selv[p], seli[p], n, lane);
    int pg = p0 + p;
    for (int j = lane; j < TOPK; j += 32) {
      topval[(size_t)pg * 64 + j] = selv[p][j];
      topidx[(size_t)pg * 64 + j] = seli[p][j];
    }
  }
}

// ---------------------------------------------------------------------------
// K1 (fallback, self-contained): on-the-fly conversion via LDS staging.
// Used only when ws_size cannot hold the preconverted B.
// ---------------------------------------------------------------------------
__global__ void __launch_bounds__(256)
s2r_gemm_topk(const float* __restrict__ A,
              const float* __restrict__ B,
              float* __restrict__ topval,
              int*   __restrict__ topidx) {
  __shared__ alignas(32) unsigned short Afrag[2][KT_N][512];
  __shared__ alignas(16) float Bbuf[TN * 32];
  __shared__ float thresh[TP];
  __shared__ int   cnt[TP];
  __shared__ float cval[TP][CAND];
  __shared__ int   cidx[TP][CAND];
  __shared__ float selv[TP][TOPK];
  __shared__ int   seli[TP][TOPK];

  const int t    = threadIdx.x;
  const int lane = t & 31;
  const int w    = t >> 5;
  const int p0   = blockIdx.x * TP;

  if (t < TP) { thresh[t] = NEG_BIG; cnt[t] = 0; }

  {
    const int e0    = w * 2;
    const int row   = lane & 15;
    const int klane = (lane & 16) ? 8 : 0;
    for (int kt = 0; kt < KT_N; ++kt) {
      #pragma unroll
      for (int ee = 0; ee < 2; ++ee) {
        int e = e0 + ee;
        int v = e >> 1, pos = e & 1;
        int K = ((v & 4) ? 16 : 0) + klane + ((v & 3) << 1) + pos;
        float f = A[(size_t)(p0 + row) * DIM + kt * 32 + K];
        unsigned short hi = f2bf(f);
        unsigned short lo = f2bf(f - bf2f(hi));
        Afrag[0][kt][lane * 16 + e] = hi;
        Afrag[1][kt][lane * 16 + e] = lo;
      }
    }
  }
  __syncthreads();

  for (int n0 = 0; n0 < M_TOTAL; n0 += TN) {
    v8f acc = {};
    for (int kt = 0; kt < KT_N; ++kt) {
      __syncthreads();
      {
        int r = t >> 1, cb = (t & 1) * 16;
        const float4* src = (const float4*)&B[(size_t)(n0 + r) * DIM + kt * 32 + cb];
        float4* dst = (float4*)&Bbuf[r * 32 + cb];
        dst[0] = src[0]; dst[1] = src[1]; dst[2] = src[2]; dst[3] = src[3];
      }
      __syncthreads();

      u16x16 ah16 = *(const u16x16*)&Afrag[0][kt][lane * 16];
      u16x16 al16 = *(const u16x16*)&Afrag[1][kt][lane * 16];
      u16x16 bh16, bl16;
      {
        const float* brow = &Bbuf[((w << 4) + (lane & 15)) * 32];
        const int klane = (lane & 16) ? 8 : 0;
        #pragma unroll
        for (int v = 0; v < 8; ++v) {
          int k = ((v & 4) ? 16 : 0) + klane + ((v & 3) << 1);
          float fx = brow[k], fy = brow[k + 1];
          unsigned short h0 = f2bf(fx), h1 = f2bf(fy);
          bh16[2 * v]     = h0;
          bh16[2 * v + 1] = h1;
          bl16[2 * v]     = f2bf(fx - bf2f(h0));
          bl16[2 * v + 1] = f2bf(fy - bf2f(h1));
        }
      }
      v16bf ah = __builtin_bit_cast(v16bf, ah16);
      v16bf al = __builtin_bit_cast(v16bf, al16);
      v16bf bh = __builtin_bit_cast(v16bf, bh16);
      v16bf bl = __builtin_bit_cast(v16bf, bl16);
      acc = __builtin_amdgcn_wmma_f32_16x16x32_bf16(false, ah, false, bh, (short)0, acc, false, false);
      acc = __builtin_amdgcn_wmma_f32_16x16x32_bf16(false, ah, false, bl, (short)0, acc, false, false);
      acc = __builtin_amdgcn_wmma_f32_16x16x32_bf16(false, al, false, bh, (short)0, acc, false, false);
    }

    const int node  = n0 + (w << 4) + (lane & 15);
    const int pbase = (lane & 16) ? 8 : 0;
    #pragma unroll
    for (int i = 0; i < 8; ++i) {
      float v = acc[i];
      int p = pbase + i;
      if (v > thresh[p]) {
        int pos = atomicAdd(&cnt[p], 1);
        if (pos < CAND) { cval[p][pos] = v; cidx[p][pos] = node; }
      }
    }
    __syncthreads();
    #pragma unroll
    for (int o = 0; o < 2; ++o) {
      int p = w + o * 8;
      int c = cnt[p];
      if (c > CAND - TN) {
        int n = c < CAND ? c : CAND;
        extract_top50(cval[p], cidx[p], selv[p], seli[p], n, lane);
        for (int j = lane; j < TOPK; j += 32) {
          cval[p][j] = selv[p][j]; cidx[p][j] = seli[p][j];
        }
        if (lane == 0) { cnt[p] = TOPK; thresh[p] = selv[p][TOPK - 1]; }
      }
    }
    __syncthreads();
  }

  #pragma unroll
  for (int o = 0; o < 2; ++o) {
    int p = w + o * 8;
    int c = cnt[p]; int n = c < CAND ? c : CAND;
    extract_top50(cval[p], cidx[p], selv[p], seli[p], n, lane);
    int pg = p0 + p;
    for (int j = lane; j < TOPK; j += 32) {
      topval[(size_t)pg * 64 + j] = selv[p][j];
      topidx[(size_t)pg * 64 + j] = seli[p][j];
    }
  }
}

__global__ void __launch_bounds__(256)
s2r_softmax_gather(const float* __restrict__ B,
                   const float* __restrict__ topval,
                   const int*   __restrict__ topidx,
                   float* __restrict__ out) {
  __shared__ float attn[TOPK];
  __shared__ int   sidx[TOPK];
  __shared__ float vals[TOPK];
  const int p = blockIdx.x, t = threadIdx.x;

  if (t < TOPK) {
    vals[t] = topval[(size_t)p * 64 + t];
    sidx[t] = topidx[(size_t)p * 64 + t];
  }
  __syncthreads();
  if (t == 0) {
    float m = vals[0];
    for (int k = 1; k < TOPK; ++k) m = fmaxf(m, vals[k]);
    float s = 0.f;
    for (int k = 0; k < TOPK; ++k) {
      float e = expf((vals[k] - m) * (1.0f / TAU));
      attn[k] = e; s += e;
    }
    float inv = 1.0f / s;
    for (int k = 0; k < TOPK; ++k) attn[k] *= inv;
  }
  __syncthreads();

  for (int d = t; d < DIM; d += 256) {
    float acc = 0.f;
    #pragma unroll 5
    for (int k = 0; k < TOPK; ++k)
      acc = fmaf(attn[k], B[(size_t)sidx[k] * DIM + d], acc);
    out[(size_t)p * DIM + d] = acc;
  }
}

extern "C" void kernel_launch(void* const* d_in, const int* in_sizes, int n_in,
                              void* d_out, int out_size, void* d_ws, size_t ws_size,
                              hipStream_t stream) {
  const float* test_patches = (const float*)d_in[0];   // [4096 x 768] f32
  const float* memory_nodes = (const float*)d_in[1];   // [65536 x 768] f32
  float* out = (float*)d_out;                          // [4096 x 768] f32
  (void)in_sizes; (void)n_in; (void)out_size;

  const size_t topk_bytes = (size_t)P_TOTAL * 64 * (sizeof(float) + sizeof(int)); // 2 MB
  const size_t frag_bytes = (size_t)(M_TOTAL / 16) * KT_N * 32 * 32;              // 100.7 MB each

  float* topval = (float*)d_ws;
  int*   topidx = (int*)((char*)d_ws + (size_t)P_TOTAL * 64 * sizeof(float));

  if (ws_size >= topk_bytes + 2 * frag_bytes) {
    // Fast path: preconvert B once into fragment-major bf16 hi/lo.
    char* BhiB = (char*)d_ws + topk_bytes;
    char* BloB = (char*)d_ws + topk_bytes + frag_bytes;
    s2r_convert_b<<<M_TOTAL / 16, 256, 0, stream>>>(memory_nodes,
        (unsigned int*)BhiB, (unsigned int*)BloB);
    s2r_gemm_topk_fast<<<P_TOTAL / TP, 256, 0, stream>>>(
        test_patches, BhiB, BloB, topval, topidx);
  } else {
    // Fallback: self-contained, converts B on the fly inside the GEMM.
    s2r_gemm_topk<<<P_TOTAL / TP, 256, 0, stream>>>(
        test_patches, memory_nodes, topval, topidx);
  }
  s2r_softmax_gather<<<P_TOTAL, 256, 0, stream>>>(memory_nodes, topval, topidx, out);
}
